// EdgeUpdate_60885456388953
// MI455X (gfx1250) — matrix operand order repaired and checked by reading, not compiled, not run on hardware
//
#include <hip/hip_runtime.h>

// ---------------------------------------------------------------------------
// EdgeUpdate MLP for gfx1250 (MI455X): bf16 WMMA (v_wmma_f32_16x16x32_bf16)
// 8 waves = 2 groups x 4 waves; group owns 32 edges (2 M-tiles),
// wave owns a 64-wide N quarter => each B fragment feeds 2 WMMAs.
// ---------------------------------------------------------------------------

typedef __bf16 v16bf __attribute__((ext_vector_type(16)));
typedef __bf16 v8bf  __attribute__((ext_vector_type(8)));
typedef float  v8f   __attribute__((ext_vector_type(8)));

#define IN_DIM      192
#define HID         256
#define BLOCK_EDGES 64                 // 2 groups x 32 edges
#define W1F_OFF     0                  // 6*16*32*16 bf16  = 96 KB
#define W2F_OFF     98304              // 8*16*32*16 bf16  = 128 KB
#define GRP_OFF     229376
#define GRP_STRIDE  28672              // 12 KB x-tile + 16 KB h-tile
#define LDS_TOTAL   286720             // 280 KB <= 320 KB per WGP

static __device__ __forceinline__ __bf16 f2bf(float f) {
    union { float f; unsigned u; } c; c.f = f;
    unsigned u = c.u + 0x7FFFu + ((c.u >> 16) & 1u);   // round-to-nearest-even
    unsigned short h = (unsigned short)(u >> 16);
    __bf16 r; __builtin_memcpy(&r, &h, 2);
    return r;
}

static __device__ __forceinline__ unsigned pack2bf(float a, float b) {
    union { float f; unsigned u; } ca, cb; ca.f = a; cb.f = b;
    unsigned ua = ca.u + 0x7FFFu + ((ca.u >> 16) & 1u);
    unsigned ub = cb.u + 0x7FFFu + ((cb.u >> 16) & 1u);
    return (ua >> 16) | (ub & 0xFFFF0000u);
}

static __device__ __forceinline__ v16bf cat16(v8bf lo, v8bf hi) {
    return __builtin_shufflevector(lo, hi, 0,1,2,3,4,5,6,7,8,9,10,11,12,13,14,15);
}

static __device__ __forceinline__ v16bf ldfragA(const __bf16* p) {
    // A-frag: lane holds row (lane&15); K = 16*(lane>>4) + e, 32 contiguous bytes
    return cat16(*(const v8bf*)p, *(const v8bf*)(p + 8));
}
static __device__ __forceinline__ v16bf ldfragB(const __bf16* p) {
    // pre-swizzled B-frag: 32 contiguous bytes per lane
    return cat16(*(const v8bf*)p, *(const v8bf*)(p + 8));
}

__global__ void __launch_bounds__(256, 1)
edge_mlp_kernel(const float* __restrict__ E, const float* __restrict__ V,
                const int* __restrict__ eidx, const float* __restrict__ W1,
                const float* __restrict__ b1, const float* __restrict__ W2,
                const float* __restrict__ b2, float* __restrict__ out,
                int nEdges, int nBlocks)
{
    extern __shared__ char smem[];
    __bf16* w1f = (__bf16*)(smem + W1F_OFF);
    __bf16* w2f = (__bf16*)(smem + W2F_OFF);

    const int t    = threadIdx.x;
    const int lane = t & 31;
    const int wave = t >> 5;
    const int g    = wave >> 2;        // group id (0,1)
    const int sub  = wave & 3;         // N-quarter id (0..3)
    const int tg   = t & 127;          // thread id within group
    const int mrow = lane & 15;
    const int hsel = lane >> 4;
    const int ncol = lane & 15;

    __bf16*   xg   = (__bf16*)(smem + GRP_OFF + g * GRP_STRIDE); // 32x192 bf16
    __bf16*   hg   = xg + 32 * IN_DIM;                           // 32x256 bf16
    unsigned* xg32 = (unsigned*)xg;

    // ---- per-lane bias registers for this wave's 4 N-tiles ----------------
    float b1r[4], b2r[4];
#pragma unroll
    for (int j = 0; j < 4; ++j) {
        b1r[j] = b1[(sub * 4 + j) * 16 + ncol];
        b2r[j] = b2[(sub * 4 + j) * 16 + ncol];
    }

    // ---- stage W1, W2 into LDS as bf16 WMMA-B fragments (once per WG) -----
    // elem index = ((kc*16 + nt)*32 + lane)*16 + e ; lane: n = lane&15,
    // K = 16*(lane>>4) + e
    for (int k = 0; k < IN_DIM; ++k) {          // coalesced global reads
        float w = W1[k * HID + t];
        int kc = k >> 5, kr = k & 31;
        int fl = ((kc * 16 + (t >> 4)) * 32 + ((kr >> 4) * 16 + (t & 15))) * 16 + (kr & 15);
        w1f[fl] = f2bf(w);
    }
    for (int k = 0; k < HID; ++k) {
        float w = W2[k * HID + t];
        int kc = k >> 5, kr = k & 31;
        int fl = ((kc * 16 + (t >> 4)) * 32 + ((kr >> 4) * 16 + (t & 15))) * 16 + (kr & 15);
        w2f[fl] = f2bf(w);
    }
    __syncthreads();

    // ---- persistent loop over 64-edge blocks ------------------------------
    for (int blk = blockIdx.x; blk < nBlocks; blk += gridDim.x) {
        const int gbase = blk * BLOCK_EDGES + g * 32;   // group's first edge

        { // prefetch next block's E rows
            int nb = blk + gridDim.x;
            if (nb < nBlocks) {
                long long pe = (long long)nb * BLOCK_EDGES + g * 32;
                __builtin_prefetch(E + pe * 64 + tg * 16, 0, 1);
            }
        }

        // ---- stage x = [E | V[src] | V[dst]] bf16 32x192, group-cooperative
        for (int i = 0; i < 24; ++i) {
            int flat = i * 128 + tg;           // 32 edges * 96 float2-pairs
            int er = flat / 96;
            int pc = flat - er * 96;
            int e  = gbase + er;
            float2 v; v.x = 0.f; v.y = 0.f;
            if (e < nEdges) {
                int col = pc * 2;
                const float* p;
                if (col < 64)        p = E + (long long)e * 64 + col;
                else if (col < 128)  p = V + (long long)eidx[e] * 64 + (col - 64);
                else                 p = V + (long long)eidx[nEdges + e] * 64 + (col - 128);
                v = *(const float2*)p;
            }
            xg32[er * 96 + pc] = pack2bf(v.x, v.y);
        }
        __syncthreads();

        // ---- GEMM1: h[32 x 64q] = x[32x192] @ W1[:, quarter] --------------
        v8f acc[2][4];
#pragma unroll
        for (int mt = 0; mt < 2; ++mt)
#pragma unroll
            for (int j = 0; j < 4; ++j) acc[mt][j] = (v8f)(0.0f);

#pragma unroll
        for (int kc = 0; kc < 6; ++kc) {
            v16bf a0 = ldfragA(xg + (mrow)      * IN_DIM + kc * 32 + hsel * 16);
            v16bf a1 = ldfragA(xg + (mrow + 16) * IN_DIM + kc * 32 + hsel * 16);
#pragma unroll
            for (int j = 0; j < 4; ++j) {
                int nt = sub * 4 + j;
                v16bf b = ldfragB(w1f + (((kc * 16 + nt) * 32) + lane) * 16);
                acc[0][j] = __builtin_amdgcn_wmma_f32_16x16x32_bf16(
                    false, a0, false, b, (short)0, acc[0][j], false, false);
                acc[1][j] = __builtin_amdgcn_wmma_f32_16x16x32_bf16(
                    false, a1, false, b, (short)0, acc[1][j], false, false);
            }
        }

        // ---- bias + ReLU, store h quarter as bf16 into group h-tile -------
#pragma unroll
        for (int mt = 0; mt < 2; ++mt)
#pragma unroll
            for (int j = 0; j < 4; ++j)
#pragma unroll
                for (int r = 0; r < 8; ++r) {
                    float hv = acc[mt][j][r] + b1r[j];
                    hv = hv > 0.0f ? hv : 0.0f;
                    int row = mt * 16 + r + 8 * hsel;
                    hg[row * HID + (sub * 4 + j) * 16 + ncol] = f2bf(hv);
                }
        __syncthreads();

        // ---- GEMM2: out[32 x 64q] = h[32x256] @ W2[:, quarter] ------------
        v8f oac[2][4];
#pragma unroll
        for (int mt = 0; mt < 2; ++mt)
#pragma unroll
            for (int j = 0; j < 4; ++j) oac[mt][j] = (v8f)(0.0f);

#pragma unroll
        for (int kc = 0; kc < 8; ++kc) {
            v16bf a0 = ldfragA(hg + (mrow)      * HID + kc * 32 + hsel * 16);
            v16bf a1 = ldfragA(hg + (mrow + 16) * HID + kc * 32 + hsel * 16);
#pragma unroll
            for (int j = 0; j < 4; ++j) {
                int nt = sub * 4 + j;
                v16bf b = ldfragB(w2f + (((kc * 16 + nt) * 32) + lane) * 16);
                oac[0][j] = __builtin_amdgcn_wmma_f32_16x16x32_bf16(
                    false, a0, false, b, (short)0, oac[0][j], false, false);
                oac[1][j] = __builtin_amdgcn_wmma_f32_16x16x32_bf16(
                    false, a1, false, b, (short)0, oac[1][j], false, false);
            }
        }

        // ---- add b2, store output (64B-contiguous per half-wave) ----------
#pragma unroll
        for (int mt = 0; mt < 2; ++mt)
#pragma unroll
            for (int j = 0; j < 4; ++j)
#pragma unroll
                for (int r = 0; r < 8; ++r) {
                    int row = mt * 16 + r + 8 * hsel;
                    int e   = gbase + row;
                    if (e < nEdges)
                        out[(long long)e * HID + (sub * 4 + j) * 16 + ncol] =
                            oac[mt][j][r] + b2r[j];
                }
    }
}

extern "C" void kernel_launch(void* const* d_in, const int* in_sizes, int n_in,
                              void* d_out, int out_size, void* d_ws, size_t ws_size,
                              hipStream_t stream) {
    const float* E    = (const float*)d_in[0];
    const float* V    = (const float*)d_in[1];
    const int*   eidx = (const int*)  d_in[2];   // [2, N_EDGES]
    const float* W1   = (const float*)d_in[3];   // [192, 256]
    const float* b1   = (const float*)d_in[4];
    const float* W2   = (const float*)d_in[5];   // [256, 256]
    const float* b2   = (const float*)d_in[6];
    float* out = (float*)d_out;

    const int nEdges  = in_sizes[0] / 64;
    const int nBlocks = (nEdges + BLOCK_EDGES - 1) / BLOCK_EDGES;

    (void)d_ws; (void)ws_size; (void)n_in; (void)out_size;

    hipFuncSetAttribute((const void*)edge_mlp_kernel,
                        hipFuncAttributeMaxDynamicSharedMemorySize, LDS_TOTAL);

    int grid = nBlocks < 2048 ? nBlocks : 2048;
    edge_mlp_kernel<<<grid, 256, LDS_TOTAL, stream>>>(
        E, V, eidx, W1, b1, W2, b2, out, nEdges, nBlocks);
}
